// Generator_58737972740271
// MI455X (gfx1250) — matrix operand order repaired and checked by reading, not compiled, not run on hardware
//
#include <hip/hip_runtime.h>

typedef __attribute__((ext_vector_type(16))) _Float16 v16h;
typedef __attribute__((ext_vector_type(8)))  float    v8f;

#define WMMA_F32_16x16x32_F16(A, B, C) \
  __builtin_amdgcn_wmma_f32_16x16x32_f16(false, (A), false, (B), (short)0, (C), false, false)

namespace {
constexpr int Bb = 32, Nn = 1024, F0 = 32, F1 = 64, F2 = 128;
constexpr int Mrows = Bb * Nn;            // 32768
constexpr float EPS = 1e-5f;
}

// Async global->LDS copy (CDNA5; ASYNCcnt-tracked). lds = byte offset in LDS
// (low 32 bits of a generic pointer to __shared__), g = 64-bit global address.
__device__ __forceinline__ void async_copy_b128(unsigned lds, unsigned long long g) {
  asm volatile("global_load_async_to_lds_b128 %0, %1, off"
               :: "v"(lds), "v"(g) : "memory");
}
__device__ __forceinline__ unsigned lds_addr(const void* p) {
  return (unsigned)(unsigned long long)p;   // generic->LDS offset (addr[31:0])
}

// ---------------- generic pointwise linear: y = act(x @ W^T + b) ----------------
__global__ void lin_kernel(const float* __restrict__ x, const float* __restrict__ W,
                           const float* __restrict__ bias, float* __restrict__ y,
                           int M, int K, int O, int act)
{
  int idx = blockIdx.x * blockDim.x + threadIdx.x;
  if (idx >= M * O) return;
  int m = idx / O, o = idx - m * O;
  const float* xr = x + (size_t)m * K;
  const float* wr = W + (size_t)o * K;
  float s = bias[o];
  #pragma unroll 8
  for (int k = 0; k < K; ++k) s = fmaf(xr[k], wr[k], s);
  if (act == 1)      s = fmaxf(s, 0.f);
  else if (act == 2) s = 1.f / (1.f + __expf(-s));
  y[idx] = s;
}

// ---------------- f32 -> f16 convert ----------------
__global__ void cvt_f16_kernel(const float* __restrict__ x, _Float16* __restrict__ y, int n)
{
  int idx = blockIdx.x * blockDim.x + threadIdx.x;
  if (idx < n) y[idx] = (_Float16)x[idx];
}

// ---------------- BatchNorm1d over channel dim N (stats over B,F), fused ReLU ----------------
__global__ __launch_bounds__(256) void bn_relu_kernel(float* __restrict__ t,
    const float* __restrict__ gam, const float* __restrict__ bet, int F)
{
  __shared__ float rs[256], rss[256];
  const int n = blockIdx.x;          // channel 0..1023
  const int tid = threadIdx.x;
  const int cnt = Bb * F;
  float s = 0.f, ss = 0.f;
  for (int i = tid; i < cnt; i += 256) {
    int b = i / F, f = i - b * F;
    float v = t[((size_t)b * Nn + n) * F + f];
    s += v; ss += v * v;
  }
  rs[tid] = s; rss[tid] = ss;
  __syncthreads();
  for (int st = 128; st > 0; st >>= 1) {
    if (tid < st) { rs[tid] += rs[tid + st]; rss[tid] += rss[tid + st]; }
    __syncthreads();
  }
  const float mean = rs[0] / (float)cnt;
  const float var  = rss[0] / (float)cnt - mean * mean;   // biased variance
  const float gg   = gam[n] * rsqrtf(var + EPS);
  const float bb   = bet[n];
  for (int i = tid; i < cnt; i += 256) {
    int b = i / F, f = i - b * F;
    size_t id = ((size_t)b * Nn + n) * F + f;
    t[id] = fmaxf((t[id] - mean) * gg + bb, 0.f);
  }
}

// ---------------- unary: Ut[b][o][n] = relu(x[b,n,:] . W[o,:] + b[o]), f16 TRANSPOSED ----------------
__global__ void unary_t_kernel(const float* __restrict__ x, const float* __restrict__ W,
                               const float* __restrict__ bias, _Float16* __restrict__ Ut)
{
  int idx = blockIdx.x * blockDim.x + threadIdx.x;   // (b, o, n) with n fastest
  if (idx >= Bb * F2 * Nn) return;
  int n = idx & (Nn - 1);
  int rest = idx >> 10;
  int o = rest & 127;
  int b = rest >> 7;
  const float* xr = x + ((size_t)b * Nn + n) * F2;
  const float* wr = W + (size_t)o * F2;
  float s = bias[o];
  #pragma unroll 8
  for (int k = 0; k < F2; ++k) s = fmaf(xr[k], wr[k], s);
  Ut[idx] = (_Float16)fmaxf(s, 0.f);
}

// ---------------- fused relation block ----------------
// out[b,i,:] = (wr/N) * sum_j [ psi*phi*<x_i,x_j> * (i!=j) ] * U[b,j,:]  +  x[b,i,:]
// 128 threads (4 waves); 64 i-rows per WG, j in chunks of 32, double-buffered
// async global->LDS staging; Gram + S@U both on v_wmma_f32_16x16x32_f16.
__global__ __launch_bounds__(128) void relation_kernel(
    const float* __restrict__ x, const _Float16* __restrict__ xh,
    const _Float16* __restrict__ Ut,
    const float* __restrict__ psi, const float* __restrict__ phi,
    const float* __restrict__ wr,
    float* __restrict__ out, _Float16* __restrict__ outh)
{
  constexpr int LSTR = F2 + 8;                       // 136 halves (272 B) row stride
  constexpr int JSTR = 40;                           // 40 halves (80 B) row stride
  constexpr int NT   = Nn / 32;                      // 32 j-tiles
  __shared__ __align__(16) _Float16 sXI[64 * LSTR];         // 64 i-rows of xh
  __shared__ __align__(16) _Float16 sXJ[2][32 * LSTR];      // j-tile of xh   [j][k]
  __shared__ __align__(16) _Float16 sUJT[2][128 * JSTR];    // j-tile of Ut   [f][j]
  __shared__ __align__(16) _Float16 sS[4][16 * 40];         // per-wave S bounce

  const int tid  = threadIdx.x;
  const int wave = tid >> 5;
  const int lane = tid & 31;
  const int b    = blockIdx.y;
  const int i0   = blockIdx.x * 64;

  const float*    xb  = x  + (size_t)b * Nn * F2;
  const _Float16* xhb = xh + (size_t)b * Nn * F2;
  const _Float16* utb = Ut + (size_t)b * F2 * Nn;

  // async stage of one j-tile (xh rows + Ut rows) into buffer c: 8 b128 per thread
  auto stage_j = [&](int j0, int c) {
    {
      int row = tid >> 2, q = tid & 3;               // 32 rows x 4 quarters (64B)
      unsigned l = lds_addr(&sXJ[c][0]) + row * (LSTR * 2) + q * 64;
      unsigned long long g =
          (unsigned long long)(size_t)(xhb + (size_t)(j0 + row) * F2 + q * 32);
      #pragma unroll
      for (int i = 0; i < 4; ++i) async_copy_b128(l + i * 16, g + i * 16);
    }
    {
      int f = tid;                                   // 128 f-rows x 64B each
      unsigned l = lds_addr(&sUJT[c][0]) + f * (JSTR * 2);
      unsigned long long g =
          (unsigned long long)(size_t)(utb + (size_t)f * Nn + j0);
      #pragma unroll
      for (int i = 0; i < 4; ++i) async_copy_b128(l + i * 16, g + i * 16);
    }
  };

  // prologue: stage the 64 i-rows (8 b128/thread) + first j-tile
  {
    int row = tid >> 1, hq = tid & 1;                // 64 rows x 2 halves (128B)
    unsigned l = lds_addr(&sXI[0]) + row * (LSTR * 2) + hq * 128;
    unsigned long long g =
        (unsigned long long)(size_t)(xhb + (size_t)(i0 + row) * F2 + hq * 64);
    #pragma unroll
    for (int i = 0; i < 8; ++i) async_copy_b128(l + i * 16, g + i * 16);
  }
  stage_j(0, 0);
  asm volatile("s_wait_asynccnt 0x8" ::: "memory");  // XI done (in-order), j0 in flight
  __syncthreads();

  // --- A fragments for this wave's 16 i-rows, K=128 as 4 chunks of 32 ---
  const int m   = lane & 15;
  const int kbA = (lane < 16) ? 0 : 8;               // A layout half-K base
  const int nb  = lane & 15;                         // B-fragment column
  const int kbB = (lane < 16) ? 0 : 8;               // B layout: halves = K kbB..kbB+7, 16+kbB..16+kbB+7
  v16h afrag[4];
  {
    const int arow = wave * 16 + m;
    #pragma unroll
    for (int kk = 0; kk < 4; ++kk) {
      #pragma unroll
      for (int h = 0; h < 8; ++h) {
        afrag[kk][h]     = sXI[arow * LSTR + kk * 32 + kbA + h];
        afrag[kk][8 + h] = sXI[arow * LSTR + kk * 32 + 16 + kbA + h];
      }
    }
  }

  const float sscale = psi[0] * phi[0];
  const int   mb = (lane < 16) ? 0 : 8;              // C/D layout: VGPR r -> row mb+r
  const int   cn = lane & 15;                        // C/D layout: column
  v8f acc[8] = {};                                   // 16x128 f32 accumulator

  for (int jt = 0; jt < NT; ++jt) {
    const int c  = jt & 1;
    const int j0 = jt * 32;
    if (jt + 1 < NT) {                               // prefetch next tile, then drain cur
      stage_j((jt + 1) * 32, c ^ 1);
      asm volatile("s_wait_asynccnt 0x8" ::: "memory");
    } else {
      asm volatile("s_wait_asynccnt 0x0" ::: "memory");
    }
    __syncthreads();                                 // all waves' tile-c data in LDS

    // ---- Gram: two 16x16 S tiles, K=128 in 4 WMMA steps each ----
    v8f s0 = {}, s1 = {};
    #pragma unroll
    for (int kk = 0; kk < 4; ++kk) {
      v16h b0, b1;
      #pragma unroll
      for (int h = 0; h < 16; ++h) {
        b0[h] = sXJ[c][nb * LSTR        + kk * 32 + kbB + h + (h & 8)];   // B[k][n]=x[j0+n][k]
        b1[h] = sXJ[c][(16 + nb) * LSTR + kk * 32 + kbB + h + (h & 8)];
      }
      s0 = WMMA_F32_16x16x32_F16(afrag[kk], b0, s0);
      s1 = WMMA_F32_16x16x32_F16(afrag[kk], b1, s1);
    }

    // scale by psi*phi, zero the i==j diagonal
    #pragma unroll
    for (int r = 0; r < 8; ++r) {
      const int gi = i0 + wave * 16 + mb + r;
      float v0 = s0[r] * sscale; if (gi == j0 + cn)      v0 = 0.f;
      float v1 = s1[r] * sscale; if (gi == j0 + 16 + cn) v1 = 0.f;
      s0[r] = v0; s1[r] = v1;
    }

    // bounce S through LDS: C-layout -> row-major [16 m][32 j] f16
    _Float16* ss = sS[wave];
    #pragma unroll
    for (int r = 0; r < 8; ++r) {
      ss[(mb + r) * 40 + cn]      = (_Float16)s0[r];
      ss[(mb + r) * 40 + 16 + cn] = (_Float16)s1[r];
    }
    __syncthreads();

    // re-fragment S as WMMA A operand (16x32, K=j)
    v16h sa;
    #pragma unroll
    for (int h = 0; h < 8; ++h) {
      sa[h]     = ss[m * 40 + kbA + h];
      sa[8 + h] = ss[m * 40 + 16 + kbA + h];
    }

    // ---- acc += S_tile @ U[j0:j0+32, :]  (8 col tiles; contiguous Ut reads) ----
    #pragma unroll
    for (int ft = 0; ft < 8; ++ft) {
      v16h bu;
      #pragma unroll
      for (int h = 0; h < 16; ++h)
        bu[h] = sUJT[c][(ft * 16 + nb) * JSTR + kbB + h + (h & 8)];       // B[k][n]=U[j0+k][f]
      acc[ft] = WMMA_F32_16x16x32_F16(sa, bu, acc[ft]);
    }
    __syncthreads();                                 // reads of tile c done
  }

  // out = (wr/N)*acc + x ; also emit f16 mirror for the next relation block
  const float wrs = wr[0] / (float)Nn;
  #pragma unroll
  for (int ft = 0; ft < 8; ++ft) {
    #pragma unroll
    for (int r = 0; r < 8; ++r) {
      const int gi = i0 + wave * 16 + mb + r;
      const int gf = ft * 16 + cn;
      const size_t id = ((size_t)b * Nn + gi) * F2 + gf;
      const float v = wrs * acc[ft][r] + xb[(size_t)gi * F2 + gf];
      out[id]  = v;
      outh[id] = (_Float16)v;
    }
  }
}

// ---------------- head: out[:, :, 0:28]=bc, [28:32]=bg ----------------
__global__ void head_kernel(const float* __restrict__ x,
                            const float* __restrict__ bcW, const float* __restrict__ bcB,
                            const float* __restrict__ bgW, const float* __restrict__ bgB,
                            float* __restrict__ out)
{
  int idx = blockIdx.x * blockDim.x + threadIdx.x;
  if (idx >= Mrows * F0) return;
  int m = idx >> 5, o = idx & 31;
  const float* wrow; float bias;
  if (o < 28) { wrow = bcW + (size_t)o * F0;        bias = bcB[o]; }
  else        { wrow = bgW + (size_t)(o - 28) * F0; bias = bgB[o - 28]; }
  const float* xr = x + (size_t)m * F0;
  float s = bias;
  #pragma unroll
  for (int k = 0; k < F0; ++k) s = fmaf(xr[k], wrow[k], s);
  out[idx] = s;
}

extern "C" void kernel_launch(void* const* d_in, const int* in_sizes, int n_in,
                              void* d_out, int out_size, void* d_ws, size_t ws_size,
                              hipStream_t stream) {
  const float* input = (const float*)d_in[0];
  const float* e1_W = (const float*)d_in[1];  const float* e1_b = (const float*)d_in[2];
  const float* bn1_g = (const float*)d_in[3]; const float* bn1_b = (const float*)d_in[4];
  const float* e2_W = (const float*)d_in[5];  const float* e2_b = (const float*)d_in[6];
  const float* bn2_g = (const float*)d_in[7]; const float* bn2_b = (const float*)d_in[8];
  const float* e3_W = (const float*)d_in[9];  const float* e3_b = (const float*)d_in[10];
  // relation params: r=0..3 at indices 11+5r .. 15+5r  (W, b, psi, phi, wr)
  const float* d1_W = (const float*)d_in[31]; const float* d1_b = (const float*)d_in[32];
  const float* dbn_g = (const float*)d_in[33]; const float* dbn_b = (const float*)d_in[34];
  const float* d2_W = (const float*)d_in[35]; const float* d2_b = (const float*)d_in[36];
  const float* bc_W = (const float*)d_in[37]; const float* bc_b = (const float*)d_in[38];
  const float* bg_W = (const float*)d_in[39]; const float* bg_b = (const float*)d_in[40];

  // ws: bufA|bufB (f32, M*128 each) | XhA|XhB (f16 mirrors) | Ut (f16 transposed unary)
  float* bufA = (float*)d_ws;
  float* bufB = bufA + (size_t)Mrows * F2;
  _Float16* XhA = (_Float16*)(bufB + (size_t)Mrows * F2);
  _Float16* XhB = XhA + (size_t)Mrows * F2;
  _Float16* Ut  = XhB + (size_t)Mrows * F2;

  const int TPB = 256;
  auto blocks = [](long long n, int t) { return (unsigned)((n + t - 1) / t); };

  // encoder
  lin_kernel<<<blocks((long long)Mrows * F1, TPB), TPB, 0, stream>>>(input, e1_W, e1_b, bufA, Mrows, F0, F1, 0);
  bn_relu_kernel<<<Nn, 256, 0, stream>>>(bufA, bn1_g, bn1_b, F1);
  lin_kernel<<<blocks((long long)Mrows * F2, TPB), TPB, 0, stream>>>(bufA, e2_W, e2_b, bufB, Mrows, F1, F2, 0);
  bn_relu_kernel<<<Nn, 256, 0, stream>>>(bufB, bn2_g, bn2_b, F2);
  lin_kernel<<<blocks((long long)Mrows * F2, TPB), TPB, 0, stream>>>(bufB, e3_W, e3_b, bufA, Mrows, F2, F2, 2);
  cvt_f16_kernel<<<blocks((long long)Mrows * F2, TPB), TPB, 0, stream>>>(bufA, XhA, Mrows * F2);

  // 4 relation blocks, ping-pong (f32 + f16 mirrors); ends back in bufA
  float* xin = bufA;   float* xout = bufB;
  _Float16* hin = XhA; _Float16* hout = XhB;
  for (int r = 0; r < 4; ++r) {
    const float* rW   = (const float*)d_in[11 + 5 * r + 0];
    const float* rb   = (const float*)d_in[11 + 5 * r + 1];
    const float* rpsi = (const float*)d_in[11 + 5 * r + 2];
    const float* rphi = (const float*)d_in[11 + 5 * r + 3];
    const float* rwr  = (const float*)d_in[11 + 5 * r + 4];
    unary_t_kernel<<<blocks((long long)Mrows * F2, TPB), TPB, 0, stream>>>(xin, rW, rb, Ut);
    relation_kernel<<<dim3(Nn / 64, Bb), 128, 0, stream>>>(xin, hin, Ut, rpsi, rphi, rwr, xout, hout);
    float* t = xin; xin = xout; xout = t;
    _Float16* th = hin; hin = hout; hout = th;
  }
  // xin == bufA here

  // decoder + heads
  lin_kernel<<<blocks((long long)Mrows * F1, TPB), TPB, 0, stream>>>(xin, d1_W, d1_b, bufB, Mrows, F2, F1, 0);
  bn_relu_kernel<<<Nn, 256, 0, stream>>>(bufB, dbn_g, dbn_b, F1);
  lin_kernel<<<blocks((long long)Mrows * F0, TPB), TPB, 0, stream>>>(bufB, d2_W, d2_b, bufA, Mrows, F1, F0, 1);
  head_kernel<<<blocks((long long)Mrows * F0, TPB), TPB, 0, stream>>>(bufA, bc_W, bc_b, bg_W, bg_b, (float*)d_out);
}
